// SpatialLocalMultiheadAttention_55911884259739
// MI455X (gfx1250) — matrix operand order repaired and checked by reading, not compiled, not run on hardware
//
#include <hip/hip_runtime.h>

// ---------------------------------------------------------------------------
// Fused spatial-local window MHA for MI455X (gfx1250, wave32, WMMA).
//
// All GEMMs run on v_wmma_f32_16x16x32_bf16 (fp32 accumulate). bf16 chosen
// because the reference is fp32 (bf16 preserves exponent range) and 16-bit
// WMMA is the highest-throughput dense path with fp32-compatible inputs.
// One workgroup (= 16 wave32, 512 threads) processes one 64-token window
// entirely in LDS; weights stream from L2 (2 MB total, resident).
// ---------------------------------------------------------------------------

typedef unsigned short ushort_t;
typedef __attribute__((ext_vector_type(16))) __bf16        bf16x16;
typedef __attribute__((ext_vector_type(8)))  float         f32x8;
typedef __attribute__((ext_vector_type(4)))  float         f32x4;
typedef __attribute__((ext_vector_type(4)))  unsigned int  u32x4;

union Frag { u32x4 q[2]; bf16x16 v; };

// Tile geometry / LDS strides (all row strides multiple of 8 bf16 -> rows
// stay 16B-aligned for ds_load_b128; stride choices are bank-conflict-free
// for 16-row fragment reads: 520*2B=260 dw (+4 banks/row), 72*2B=36 dw).
#define LDX   520   // 64 x C bf16 tiles (qkin / xw / attn_out)
#define LDH   72    // 64 x 64 per-head bf16 tiles (Qh / Kh / Vt / P)
#define LDS_S 68    // 64 x 64 f32 score tile

__device__ __forceinline__ unsigned short f2bf(float x) {
  unsigned int u = __float_as_uint(x);
  u += 0x7FFFu + ((u >> 16) & 1u);          // round-to-nearest-even
  return (unsigned short)(u >> 16);
}
__device__ __forceinline__ unsigned int pack2bf(float a, float b) {
  return (unsigned int)f2bf(a) | ((unsigned int)f2bf(b) << 16);
}

// A fragment (16x32 bf16, MxK): lane m = lane&15, klo = (lane>>4)*8.
// VGPRs 0..3 <- K [klo, klo+8), VGPRs 4..7 <- K [16+klo, 16+klo+8).
__device__ __forceinline__ bf16x16 load_a_frag(const ushort_t* p0, int ld, int lane) {
  const ushort_t* p = p0 + (lane & 15) * ld + ((lane >> 4) << 3);
  Frag f;
  f.q[0] = *(const u32x4*)(p);
  f.q[1] = *(const u32x4*)(p + 16);
  return f.v;
}

// B fragment (32x16 bf16, KxN) read from memory holding B^T row-major [N][K]:
// lane n = lane&15, klo = (lane>>4)*8: K [klo, klo+8) then [16+klo, 16+klo+8).
// (Weights W[out][in] are exactly B^T for X @ W^T, so they stream directly.)
__device__ __forceinline__ bf16x16 load_b_frag(const ushort_t* p0, int ld, int lane) {
  const ushort_t* p = p0 + (lane & 15) * ld + ((lane >> 4) << 3);
  Frag f;
  f.q[0] = *(const u32x4*)(p);
  f.q[1] = *(const u32x4*)(p + 16);
  return f.v;
}

__device__ __forceinline__ f32x8 wmma_bf16(bf16x16 a, bf16x16 b, f32x8 c) {
  return __builtin_amdgcn_wmma_f32_16x16x32_bf16(
      /*neg_a=*/false, a, /*neg_b=*/false, b,
      /*c_mod=*/(short)0, c, /*reuse_a=*/false, /*reuse_b=*/false);
}

#define SMEM_BYTES ((3 * 64 * LDX + 4 * 64 * LDH) * 2 + 64 * LDS_S * 4)  // 253952

// ---------------------------------------------------------------------------
// Weight fp32 -> bf16 conversion (runs every call; deterministic, ~2 MB).
// ---------------------------------------------------------------------------
__global__ void wcvt_kernel(const float* __restrict__ wi,   // (1536,512)
                            const float* __restrict__ wo,   // (512,512)
                            ushort_t* __restrict__ dst) {
  int i = blockIdx.x * blockDim.x + threadIdx.x;
  if (i < 786432)        dst[i] = f2bf(wi[i]);
  else if (i < 1048576)  dst[i] = f2bf(wo[i - 786432]);
}

// ---------------------------------------------------------------------------
// Fused window attention. grid = 2048 windows, block = 512 threads.
// ---------------------------------------------------------------------------
__global__ __launch_bounds__(512)
void slmha_fused_kernel(const float* __restrict__ x,        // (N,T,64,64,512)
                        const float* __restrict__ pos,      // (8,8,512)
                        const float* __restrict__ in_bias,  // (1536)
                        const float* __restrict__ out_bias, // (512)
                        const ushort_t* __restrict__ wqkv,  // (1536,512) bf16
                        const ushort_t* __restrict__ wo,    // (512,512) bf16
                        float* __restrict__ out) {
  extern __shared__ char smem_raw[];
  ushort_t* qkS = (ushort_t*)smem_raw;        // xw + pos (Q/K input), 64 x LDX
  ushort_t* xwS = qkS + 64 * LDX;             // xw (V input),        64 x LDX
  ushort_t* aoS = xwS + 64 * LDX;             // attention output,    64 x LDX
  ushort_t* QhS = aoS + 64 * LDX;             // Q_h [tok][d],  64 x LDH
  ushort_t* KhS = QhS + 64 * LDH;             // K_h [tok][d],  64 x LDH
  ushort_t* VtS = KhS + 64 * LDH;             // V_h [d][tok],  64 x LDH
  ushort_t* PhS = VtS + 64 * LDH;             // P   [tok][tok],64 x LDH
  float*    Sb  = (float*)(PhS + 64 * LDH);   // scores f32,    64 x LDS_S

  const int tid  = threadIdx.x;
  const int lane = tid & 31;
  const int wv   = tid >> 5;
  const int half = lane >> 4;
  const int nlo  = lane & 15;

  const int b  = blockIdx.x;      // window id
  const int nt = b >> 6;          // N*T slice
  const int wi = (b >> 3) & 7;    // window row
  const int wj = b & 7;           // window col
  const long xbase = (long)nt * (64 * 64 * 512);

  // ---- Stage 1: gather window, add pos-embed, fp32 -> bf16 into LDS ----
  for (int idx = tid; idx < 64 * 128; idx += 512) {        // 64 rows x 128 f32x4
    const int l  = idx >> 7;
    const int c  = (idx & 127) << 2;
    const int wy = l >> 3, wx = l & 7;
    const long go = xbase + (long)(((wi * 8 + wy) * 64) + (wj * 8 + wx)) * 512 + c;
    const f32x4 xv = *(const f32x4*)(x + go);
    const f32x4 pv = *(const f32x4*)(pos + l * 512 + c);
    unsigned int* xd = (unsigned int*)(xwS + l * LDX + c);
    unsigned int* qd = (unsigned int*)(qkS + l * LDX + c);
    xd[0] = pack2bf(xv.x, xv.y);
    xd[1] = pack2bf(xv.z, xv.w);
    qd[0] = pack2bf(xv.x + pv.x, xv.y + pv.y);
    qd[1] = pack2bf(xv.z + pv.z, xv.w + pv.w);
  }
  __syncthreads();

  // ---- Stage 2: per-head QKV projection + attention ----
  for (int h = 0; h < 8; ++h) {
    // (a) Q_h, K_h, V_h: 48 output tiles (16x16), K=512 -> 16 wmma/tile.
    for (int s = 0; s < 3; ++s) {
      const int job   = wv + (s << 4);
      const int which = job >> 4;            // 0=Q 1=K 2=V
      const int tile  = job & 15;
      const int tm = tile >> 2, tn = tile & 3;
      const ushort_t* Asrc = (which == 2) ? xwS : qkS;
      const ushort_t* Wb = wqkv + (long)((which << 9) + (h << 6) + (tn << 4)) * 512;
      f32x8 acc = {0.f, 0.f, 0.f, 0.f, 0.f, 0.f, 0.f, 0.f};
#pragma unroll 4
      for (int kt = 0; kt < 16; ++kt) {
        if (kt < 12) __builtin_prefetch(Wb + ((kt + 4) << 5), 0, 0); // global_prefetch_b8
        const bf16x16 a  = load_a_frag(Asrc + (tm << 4) * LDX + (kt << 5), LDX, lane);
        const bf16x16 bb = load_b_frag(Wb + (kt << 5), 512, lane);
        acc = wmma_bf16(a, bb, acc);
      }
      const float bias = in_bias[(which << 9) + (h << 6) + (tn << 4) + nlo];
      if (which == 2) {                        // V stored transposed [d][token]
        ushort_t* p = VtS + ((tn << 4) + nlo) * LDH + (tm << 4) + (half << 3);
#pragma unroll
        for (int r = 0; r < 8; ++r) p[r] = f2bf(acc[r] + bias);
      } else {                                 // Q gets hd^-0.5 = 0.125 folded in
        const float scl = (which == 0) ? 0.125f : 1.0f;
        ushort_t* dst = (which == 0) ? QhS : KhS;
        ushort_t* p = dst + ((tm << 4) + (half << 3)) * LDH + (tn << 4) + nlo;
#pragma unroll
        for (int r = 0; r < 8; ++r) p[r * LDH] = f2bf((acc[r] + bias) * scl);
      }
    }
    __syncthreads();

    // (b) S = Q_h @ K_h^T  (K_h [tok][d] is already B^T layout)
    {
      const int tm = wv >> 2, tn = wv & 3;
      f32x8 acc = {0.f, 0.f, 0.f, 0.f, 0.f, 0.f, 0.f, 0.f};
#pragma unroll
      for (int kt = 0; kt < 2; ++kt) {
        const bf16x16 a  = load_a_frag(QhS + (tm << 4) * LDH + (kt << 5), LDH, lane);
        const bf16x16 bb = load_b_frag(KhS + (tn << 4) * LDH + (kt << 5), LDH, lane);
        acc = wmma_bf16(a, bb, acc);
      }
      float* p = Sb + ((tm << 4) + (half << 3)) * LDS_S + (tn << 4) + nlo;
#pragma unroll
      for (int r = 0; r < 8; ++r) p[r * LDS_S] = acc[r];
    }
    __syncthreads();

    // (c) row softmax (4 rows / wave; lane covers cols {lane, lane+32})
#pragma unroll
    for (int rr = 0; rr < 4; ++rr) {
      const int row = (wv << 2) + rr;
      const float s0 = Sb[row * LDS_S + lane];
      const float s1 = Sb[row * LDS_S + lane + 32];
      float m = fmaxf(s0, s1);
#pragma unroll
      for (int off = 16; off >= 1; off >>= 1) m = fmaxf(m, __shfl_xor(m, off, 32));
      const float e0 = __expf(s0 - m);
      const float e1 = __expf(s1 - m);
      float sum = e0 + e1;
#pragma unroll
      for (int off = 16; off >= 1; off >>= 1) sum += __shfl_xor(sum, off, 32);
      const float inv = __builtin_amdgcn_rcpf(sum);
      PhS[row * LDH + lane]      = f2bf(e0 * inv);
      PhS[row * LDH + lane + 32] = f2bf(e1 * inv);
    }
    __syncthreads();

    // (d) O_h = P @ V  (V^T [d][tok] is B^T layout) -> attn_out cols h*64..
    {
      const int tm = wv >> 2, tn = wv & 3;
      f32x8 acc = {0.f, 0.f, 0.f, 0.f, 0.f, 0.f, 0.f, 0.f};
#pragma unroll
      for (int kt = 0; kt < 2; ++kt) {
        const bf16x16 a  = load_a_frag(PhS + (tm << 4) * LDH + (kt << 5), LDH, lane);
        const bf16x16 bb = load_b_frag(VtS + (tn << 4) * LDH + (kt << 5), LDH, lane);
        acc = wmma_bf16(a, bb, acc);
      }
      ushort_t* p = aoS + ((tm << 4) + (half << 3)) * LDX + (h << 6) + (tn << 4) + nlo;
#pragma unroll
      for (int r = 0; r < 8; ++r) p[r * LDX] = f2bf(acc[r]);
    }
    __syncthreads();   // protect Qh/Kh/Vt before next head overwrites
  }

  // ---- Stage 3: out projection (64x512, K=512) + window scatter ----
  for (int s = 0; s < 8; ++s) {
    const int job = wv + (s << 4);          // 128 tiles
    const int tm = job >> 5, tn = job & 31;
    const ushort_t* Wb = wo + (long)(tn << 4) * 512;
    f32x8 acc = {0.f, 0.f, 0.f, 0.f, 0.f, 0.f, 0.f, 0.f};
#pragma unroll 4
    for (int kt = 0; kt < 16; ++kt) {
      if (kt < 12) __builtin_prefetch(Wb + ((kt + 4) << 5), 0, 0);
      const bf16x16 a  = load_a_frag(aoS + (tm << 4) * LDX + (kt << 5), LDX, lane);
      const bf16x16 bb = load_b_frag(Wb + (kt << 5), 512, lane);
      acc = wmma_bf16(a, bb, acc);
    }
    const int col = (tn << 4) + nlo;
    const float bias = out_bias[col];
#pragma unroll
    for (int r = 0; r < 8; ++r) {
      const int token = (tm << 4) + (half << 3) + r;
      const int wy = token >> 3, wx = token & 7;
      out[xbase + (long)(((wi * 8 + wy) * 64) + (wj * 8 + wx)) * 512 + col] =
          acc[r] + bias;
    }
  }
}

// ---------------------------------------------------------------------------
extern "C" void kernel_launch(void* const* d_in, const int* in_sizes, int n_in,
                              void* d_out, int out_size, void* d_ws, size_t ws_size,
                              hipStream_t stream) {
  const float* x        = (const float*)d_in[0];  // (2,16,64,64,512)
  const float* pos      = (const float*)d_in[1];  // (8,8,512)
  const float* in_w     = (const float*)d_in[2];  // (1536,512)
  const float* in_bias  = (const float*)d_in[3];  // (1536,)
  const float* out_w    = (const float*)d_in[4];  // (512,512)
  const float* out_bias = (const float*)d_in[5];  // (512,)
  float* out = (float*)d_out;

  ushort_t* wqkv_bf = (ushort_t*)d_ws;            // 786432 bf16
  ushort_t* wo_bf   = wqkv_bf + 786432;           // 262144 bf16  (2 MB total)

  wcvt_kernel<<<4096, 256, 0, stream>>>(in_w, out_w, wqkv_bf);

  slmha_fused_kernel<<<2048, 512, SMEM_BYTES, stream>>>(
      x, pos, in_bias, out_bias, wqkv_bf, wo_bf, out);
}
